// Encoder_20598663151642
// MI455X (gfx1250) — matrix operand order repaired and checked by reading, not compiled, not run on hardware
//
#include <hip/hip_runtime.h>
#include <hip/hip_bf16.h>

// ---------------------------------------------------------------------------
// Bidirectional GRU encoder for MI455X (gfx1250, wave32, WMMA).
//   V=32000, E=512, H=512, B=32, S=512
// Phase 1: gx[dir][s][b][3H] = emb[input] @ W_ih^T + b_ih  (bf16 WMMA GEMM)
// Phase 2: persistent per-direction scan workgroup, h/gh in LDS (320KB WGP),
//          W_hh (bf16, 1.5MB/dir) streamed from L2 each step, gx prefetched
//          one step ahead into cache via global_prefetch.
// ---------------------------------------------------------------------------

typedef __attribute__((ext_vector_type(16))) __bf16 v16bf;
typedef __attribute__((ext_vector_type(8)))  float  v8f;

#define Bsz   32
#define Ssz   512
#define Hsz   512
#define G3H   1536          // 3*H
#define KTOT  512           // E == H == 512
#define WELEM (G3H * KTOT)  // elements per weight matrix = 786432

static __device__ __forceinline__ __bf16 f2bf(float x) { return (__bf16)x; }

// ---------------------------------------------------------------------------
// f32 -> bf16 weight conversion
// ---------------------------------------------------------------------------
__global__ void cvt_bf16_kernel(const float* __restrict__ src,
                                __bf16* __restrict__ dst, int n) {
    int i = blockIdx.x * blockDim.x + threadIdx.x;
    if (i < n) dst[i] = f2bf(src[i]);
}

// ---------------------------------------------------------------------------
// Fragment builders (CDNA5 WMMA VGPR layouts, wave32)
// A: 16x32 bf16 (MxK). lane<16: M=lane,   K = {k0..k0+7} U {k0+16..k0+23}
//                      lane>=16: M=lane-16, K = {k0+8..k0+15} U {k0+24..k0+31}
// ---------------------------------------------------------------------------
static __device__ __forceinline__ v16bf load_a_frag(const __bf16* __restrict__ row,
                                                    int k0, int half) {
    v16bf a;
#pragma unroll
    for (int i = 0; i < 8; ++i) {
        a[i]     = row[k0 + half * 8 + i];
        a[8 + i] = row[k0 + 16 + half * 8 + i];
    }
    return a;
}

// B: 32x16 bf16 (KxN), B[k][n] = W[n][k] (W row-major [N][K]).
// lane&15 = N; lane>>4 selects K: K = {k0+h*8..+7} U {k0+16+h*8..+7} (same as A).
// Each lane reads two runs of 8 contiguous bf16 (16B) from its W row.
static __device__ __forceinline__ v16bf load_b_frag(const __bf16* __restrict__ wrow,
                                                    int k0, int half) {
    v16bf b;
    const __bf16* p = wrow + k0 + half * 8;
#pragma unroll
    for (int e = 0; e < 8; ++e) {
        b[e]     = p[e];
        b[8 + e] = p[16 + e];
    }
    return b;
}

static __device__ __forceinline__ v8f wmma_bf16(v16bf a, v16bf b, v8f c) {
    return __builtin_amdgcn_wmma_f32_16x16x32_bf16(
        false, a, false, b, (short)0, c, false, false);
}

// ---------------------------------------------------------------------------
// Phase 1: embedding gather + input projection.
// grid = (1024, 24), block = 256 (8 waves).
// blockIdx.x: 16-row M block of flattened (b,s).  blockIdx.y*128: N offset in
// the 2*1536 concatenated [fwd|bwd] column space (12 y-blocks per direction).
// ---------------------------------------------------------------------------
#define AS_STRIDE 520  // bf16 elements, padded

__global__ void embed_project_kernel(const int* __restrict__ input,
                                     const float* __restrict__ emb,
                                     const __bf16* __restrict__ Wih_bf, // [2][1536][512]
                                     const float* __restrict__ bih_f,
                                     const float* __restrict__ bih_b,
                                     float* __restrict__ gx)            // [2][512][32][1536]
{
    __shared__ __bf16 As[16 * AS_STRIDE];

    const int t = threadIdx.x;
    // --- gather 16 embedding rows into LDS as bf16 ---
    {
        const int row  = t >> 4;          // 0..15
        const int c0   = (t & 15) * 32;   // 16 threads/row * 32 cols
        const int flat = blockIdx.x * 16 + row;      // = b*S + s
        const int tok  = input[flat];
        const float* er = emb + (size_t)tok * KTOT;
        __bf16* dst = &As[row * AS_STRIDE];
#pragma unroll 8
        for (int i = 0; i < 32; ++i) dst[c0 + i] = f2bf(er[c0 + i]);
    }
    __syncthreads();

    const int wave = t >> 5;
    const int lane = t & 31;
    const int l15  = lane & 15;
    const int half = lane >> 4;

    const int nglob = blockIdx.y * 128 + wave * 16;  // 0..3071
    const int dir   = nglob / G3H;                   // 1536 % 128 == 0
    const int nd    = nglob - dir * G3H;             // tile col within direction

    const __bf16* W = Wih_bf + (size_t)dir * WELEM;
    const __bf16* wrow = W + (size_t)(nd + l15) * KTOT;  // N = lane&15
    const __bf16* arow = &As[l15 * AS_STRIDE];           // M = lane&15

    v8f acc = {};
#pragma unroll
    for (int kt = 0; kt < 16; ++kt) {
        const int k0 = kt * 32;
        v16bf a = load_a_frag(arow, k0, half);
        v16bf b = load_b_frag(wrow, k0, half);
        acc = wmma_bf16(a, b, acc);
    }

    const float bias = (dir ? bih_b : bih_f)[nd + l15];
    float* gxd = gx + (size_t)dir * (size_t)Ssz * Bsz * G3H;
    const int n_col  = nd + l15;
    const int m_base = half * 8;
#pragma unroll
    for (int r = 0; r < 8; ++r) {
        const int m    = m_base + r;
        const int flat = blockIdx.x * 16 + m;
        const int b_i  = flat >> 9;     // / S
        const int s_i  = flat & 511;    // % S
        gxd[((size_t)(s_i * Bsz + b_i)) * G3H + n_col] = acc[r] + bias;
    }
}

// ---------------------------------------------------------------------------
// Phase 2: persistent GRU scan. grid = 2 (dir), block = 1024 (32 waves).
// LDS: h f32 [32][520] + h bf16 [32][520] + gh f32 [32][1548]  (~298 KB)
// Per step: gh = h @ W_hh^T. Each wave owns N-tiles {w, w+32, w+64} and both
// M-tiles: per k-step 2 A-frags + 3 B-frags feed 6 WMMAs (max operand reuse
// on the serial critical path).
// ---------------------------------------------------------------------------
#define H_STRIDE  520
#define GH_STRIDE 1548  // not a multiple of 8 -> 8-row offset != 0 mod 64 banks
#define SCAN_LDS_BYTES (Bsz*H_STRIDE*4 + Bsz*H_STRIDE*2 + Bsz*GH_STRIDE*4)

static __device__ __forceinline__ float sigm(float x) {
    return 1.0f / (1.0f + __expf(-x));
}
static __device__ __forceinline__ float fast_tanh(float x) {
    return 1.0f - 2.0f / (__expf(2.0f * x) + 1.0f);
}

__global__ void gru_scan_kernel(const __bf16* __restrict__ Whh_bf, // [2][1536][512]
                                const float* __restrict__ gx,      // [2][512][32][1536]
                                const float* __restrict__ bhh_f,
                                const float* __restrict__ bhh_b,
                                float* __restrict__ out,           // [B][S][2H]
                                float* __restrict__ hidden)        // [1][B][2H]
{
    extern __shared__ char smem[];
    float*  hS  = (float*)smem;                                   // [32][H_STRIDE]
    __bf16* hB  = (__bf16*)(smem + Bsz * H_STRIDE * 4);           // [32][H_STRIDE]
    float*  ghS = (float*)(smem + Bsz * H_STRIDE * 4
                                + Bsz * H_STRIDE * 2);            // [32][GH_STRIDE]

    const int dir  = blockIdx.x;
    const int t    = threadIdx.x;
    const int wave = t >> 5;
    const int lane = t & 31;
    const int l15  = lane & 15;
    const int half = lane >> 4;

    const __bf16* W   = Whh_bf + (size_t)dir * WELEM;
    const float*  gxd = gx + (size_t)dir * (size_t)Ssz * Bsz * G3H;
    const float*  bhh = dir ? bhh_b : bhh_f;

    // this wave's 3 W_hh row blocks (N-tiles w, w+32, w+64) and 2 A rows
    const __bf16* wrow0 = W + (size_t)((wave +  0) * 16 + l15) * KTOT;
    const __bf16* wrow1 = W + (size_t)((wave + 32) * 16 + l15) * KTOT;
    const __bf16* wrow2 = W + (size_t)((wave + 64) * 16 + l15) * KTOT;
    const __bf16* a0row = hB + (size_t)(l15) * H_STRIDE;        // M-tile 0
    const __bf16* a1row = hB + (size_t)(16 + l15) * H_STRIDE;   // M-tile 1

    // h0 = 0
    for (int i = t; i < Bsz * H_STRIDE; i += 1024) {
        hS[i] = 0.0f;
        hB[i] = f2bf(0.0f);
    }
    __syncthreads();

    for (int step = 0; step < Ssz; ++step) {
        const int tt = dir ? (Ssz - 1 - step) : step;

        // prefetch next step's gx slab (192KB) into cache while we do WMMAs
        if (step + 1 < Ssz) {
            const int tn = dir ? (Ssz - 2 - step) : (step + 1);
            const float* pf = gxd + (size_t)tn * Bsz * G3H;
            // 32*1536*4B = 1536 cachelines of 128B
            for (int i = t; i < 1536; i += 1024)
                __builtin_prefetch(pf + (size_t)i * 32, 0, 0);
        }

        // ---- gh = h @ W_hh^T : 6 tiles/wave with A/B fragment reuse ----
        {
            v8f acc00 = {}, acc01 = {}, acc02 = {};
            v8f acc10 = {}, acc11 = {}, acc12 = {};
#pragma unroll 4
            for (int kt = 0; kt < 16; ++kt) {
                const int k0 = kt * 32;
                v16bf a0 = load_a_frag(a0row, k0, half);
                v16bf a1 = load_a_frag(a1row, k0, half);
                v16bf b0 = load_b_frag(wrow0, k0, half);
                acc00 = wmma_bf16(a0, b0, acc00);
                acc10 = wmma_bf16(a1, b0, acc10);
                v16bf b1 = load_b_frag(wrow1, k0, half);
                acc01 = wmma_bf16(a0, b1, acc01);
                acc11 = wmma_bf16(a1, b1, acc11);
                v16bf b2 = load_b_frag(wrow2, k0, half);
                acc02 = wmma_bf16(a0, b2, acc02);
                acc12 = wmma_bf16(a1, b2, acc12);
            }
            // write all 6 tiles to gh LDS
            const int r0 = half * 8;                 // C/D: M = half*8 + r
#pragma unroll
            for (int r = 0; r < 8; ++r) {
                float* g0 = ghS + (size_t)(r0 + r) * GH_STRIDE + l15;
                float* g1 = ghS + (size_t)(16 + r0 + r) * GH_STRIDE + l15;
                g0[(wave +  0) * 16] = acc00[r];
                g0[(wave + 32) * 16] = acc01[r];
                g0[(wave + 64) * 16] = acc02[r];
                g1[(wave +  0) * 16] = acc10[r];
                g1[(wave + 32) * 16] = acc11[r];
                g1[(wave + 64) * 16] = acc12[r];
            }
        }
        __syncthreads();

        // ---- gates + state update: 32*512 elements, 16 per thread ----
        const float* gxt = gxd + (size_t)tt * Bsz * G3H;
#pragma unroll
        for (int i = 0; i < 16; ++i) {
            const int idx = i * 1024 + t;
            const int b_i = idx >> 9;     // 0..31
            const int j   = idx & 511;    // 0..511
            const float* gxr = gxt + (size_t)b_i * G3H;
            const float* ghr = ghS + (size_t)b_i * GH_STRIDE;

            const float rg = sigm(gxr[j]         + ghr[j]         + bhh[j]);
            const float zg = sigm(gxr[Hsz + j]   + ghr[Hsz + j]   + bhh[Hsz + j]);
            const float ng = fast_tanh(gxr[2*Hsz + j] +
                                       rg * (ghr[2*Hsz + j] + bhh[2*Hsz + j]));
            const float hold = hS[b_i * H_STRIDE + j];
            const float hnew = (1.0f - zg) * ng + zg * hold;

            hS[b_i * H_STRIDE + j] = hnew;
            hB[b_i * H_STRIDE + j] = f2bf(hnew);
            out[((size_t)(b_i * Ssz + tt)) * (2 * Hsz) + dir * Hsz + j] = hnew;
        }
        __syncthreads();
    }

    // final hidden state [1][B][2H]
#pragma unroll
    for (int i = 0; i < 16; ++i) {
        const int idx = i * 1024 + t;
        const int b_i = idx >> 9;
        const int j   = idx & 511;
        hidden[(size_t)b_i * (2 * Hsz) + dir * Hsz + j] = hS[b_i * H_STRIDE + j];
    }
}

// ---------------------------------------------------------------------------
// Host launcher
// ---------------------------------------------------------------------------
extern "C" void kernel_launch(void* const* d_in, const int* in_sizes, int n_in,
                              void* d_out, int out_size, void* d_ws, size_t ws_size,
                              hipStream_t stream) {
    const int*   input  = (const int*)d_in[0];
    const float* emb    = (const float*)d_in[1];
    const float* W_ih_f = (const float*)d_in[2];
    const float* W_hh_f = (const float*)d_in[3];
    const float* b_ih_f = (const float*)d_in[4];
    const float* b_hh_f = (const float*)d_in[5];
    const float* W_ih_b = (const float*)d_in[6];
    const float* W_hh_b = (const float*)d_in[7];
    const float* b_ih_b = (const float*)d_in[8];
    const float* b_hh_b = (const float*)d_in[9];

    // Workspace layout:
    //   [0)        Wih_bf  [2][1536][512] bf16   (3,145,728 B)
    //   [3145728)  Whh_bf  [2][1536][512] bf16   (3,145,728 B)
    //   [6291456)  gx      [2][512][32][1536] f32 (201,326,592 B)
    char* ws = (char*)d_ws;
    __bf16* Wih_bf = (__bf16*)ws;
    __bf16* Whh_bf = (__bf16*)(ws + (size_t)2 * WELEM * sizeof(__bf16));
    float*  gx     = (float*)(ws + (size_t)4 * WELEM * sizeof(__bf16));

    // 1) convert weights to bf16
    {
        const int nblk = (WELEM + 255) / 256;
        cvt_bf16_kernel<<<nblk, 256, 0, stream>>>(W_ih_f, Wih_bf,          WELEM);
        cvt_bf16_kernel<<<nblk, 256, 0, stream>>>(W_ih_b, Wih_bf + WELEM,  WELEM);
        cvt_bf16_kernel<<<nblk, 256, 0, stream>>>(W_hh_f, Whh_bf,          WELEM);
        cvt_bf16_kernel<<<nblk, 256, 0, stream>>>(W_hh_b, Whh_bf + WELEM,  WELEM);
    }

    // 2) embedding gather + input projection (both directions)
    embed_project_kernel<<<dim3(1024, 24), 256, 0, stream>>>(
        input, emb, Wih_bf, b_ih_f, b_ih_b, gx);

    // 3) persistent GRU scan, one workgroup per direction, ~298 KB LDS each
    static_cast<void>(hipFuncSetAttribute(
        (const void*)gru_scan_kernel,
        hipFuncAttributeMaxDynamicSharedMemorySize, SCAN_LDS_BYTES));
    float* out_p    = (float*)d_out;
    float* hidden_p = out_p + (size_t)Bsz * Ssz * 2 * Hsz;  // 16,777,216
    gru_scan_kernel<<<2, 1024, SCAN_LDS_BYTES, stream>>>(
        Whh_bf, gx, b_hh_f, b_hh_b, out_p, hidden_p);
}
